// GRUCore_66718021976068
// MI455X (gfx1250) — compile-verified, matrix operand order checked by reading, not run on hardware
//
#include <hip/hip_runtime.h>
#include <hip/hip_bf16.h>

#define Tn 256
#define Bn 64
#define Hn 1024
#define H3 (3 * Hn)

typedef __bf16 bf16_t;
typedef __attribute__((ext_vector_type(16))) __bf16 v16bf;
typedef __attribute__((ext_vector_type(8)))  __bf16 v8bf;
typedef __attribute__((ext_vector_type(8)))  float  v8f;

// ---------------------------------------------------------------------------
// WMMA helper: D = A(16x32 bf16) x B(32x16 bf16) + C(16x16 f32)
// ---------------------------------------------------------------------------
__device__ __forceinline__ v8f wmma_bf16(v16bf a, v16bf b, v8f c) {
  return __builtin_amdgcn_wmma_f32_16x16x32_bf16(
      /*neg_a=*/false, a, /*neg_b=*/false, b,
      /*c_mod=*/(short)0, c, /*reuse_a=*/false, /*reuse_b=*/false);
}

// A fragment (16x32 bf16) from a row-major bf16 row pointer already offset by
// base0 = (lane>>4)*8: two contiguous 16B loads, concatenated.
__device__ __forceinline__ v16bf make_a_frag_bf(const bf16_t* __restrict__ p) {
  const v8bf lo = *(const v8bf*)(p);        // K = base0 + 0..7
  const v8bf hi = *(const v8bf*)(p + 16);   // K = base0 + 16..23
  return __builtin_shufflevector(lo, hi, 0, 1, 2, 3, 4, 5, 6, 7,
                                 8, 9, 10, 11, 12, 13, 14, 15);
}

__device__ __forceinline__ float sigf(float x)     { return 1.0f / (1.0f + __expf(-x)); }
__device__ __forceinline__ float tanhfast(float x) { return 2.0f / (1.0f + __expf(-2.0f * x)) - 1.0f; }

// ---------------------------------------------------------------------------
// Elementwise fp32 -> bf16 conversion, 8 elements per thread.
// ---------------------------------------------------------------------------
__global__ void cvt_bf16_kernel(const float* __restrict__ in, bf16_t* __restrict__ out, int n8) {
  const int i = blockIdx.x * blockDim.x + threadIdx.x;
  if (i >= n8) return;
  const float4 f0 = *(const float4*)(in + (size_t)i * 8);
  const float4 f1 = *(const float4*)(in + (size_t)i * 8 + 4);
  v8bf o;
  o[0] = (bf16_t)f0.x; o[1] = (bf16_t)f0.y; o[2] = (bf16_t)f0.z; o[3] = (bf16_t)f0.w;
  o[4] = (bf16_t)f1.x; o[5] = (bf16_t)f1.y; o[6] = (bf16_t)f1.z; o[7] = (bf16_t)f1.w;
  *(v8bf*)(out + (size_t)i * 8) = o;
}

// ---------------------------------------------------------------------------
// Swizzle Wi [H,3H] fp32 into generic WMMA B-fragment order, bf16.
// Element ((ntile*32 + kt)*32 + lane)*16 + idx ;
//   n = ntile*16 + (lane&15); kk = (idx&7)|((lane>>4)<<3)|((idx>>3)<<4); k = kt*32+kk
// ---------------------------------------------------------------------------
__global__ void swizzle_wi_kernel(const float* __restrict__ W, bf16_t* __restrict__ out) {
  const int tid = blockIdx.x * blockDim.x + threadIdx.x;
  if (tid >= 192 * 32 * 32) return;
  const int lane  = tid & 31;
  const int kt    = (tid >> 5) & 31;
  const int ntile = tid >> 10;
  const int n = ntile * 16 + (lane & 15);
  bf16_t* dst = out + (size_t)tid * 16;
#pragma unroll
  for (int idx = 0; idx < 16; ++idx) {
    const int kk = (idx & 7) | ((lane >> 4) << 3) | ((idx >> 3) << 4);
    dst[idx] = (bf16_t)W[(size_t)(kt * 32 + kk) * H3 + n];
  }
}

// ---------------------------------------------------------------------------
// Swizzle Wh [H,3H] into the scan-kernel interleaved layout:
// fragment index tid = ((c*32 + kt)*3 + g)*32 + lane, c = h-column block 0..63,
// g = gate 0..2.  Gives one contiguous 3KB block per (c,kt).
// ---------------------------------------------------------------------------
__global__ void swizzle_wh_kernel(const float* __restrict__ W, bf16_t* __restrict__ out) {
  const int tid = blockIdx.x * blockDim.x + threadIdx.x;
  if (tid >= 64 * 32 * 3 * 32) return;
  const int lane = tid & 31;
  const int g    = (tid >> 5) % 3;
  const int kt   = (tid / (32 * 3)) & 31;
  const int c    = tid / (32 * 3 * 32);
  const int n = g * Hn + c * 16 + (lane & 15);
  bf16_t* dst = out + (size_t)tid * 16;
#pragma unroll
  for (int idx = 0; idx < 16; ++idx) {
    const int kk = (idx & 7) | ((lane >> 4) << 3) | ((idx >> 3) << 4);
    dst[idx] = (bf16_t)W[(size_t)(kt * 32 + kk) * H3 + n];
  }
}

// ---------------------------------------------------------------------------
// Gi = X @ Wi + bi  (X bf16 [T*B,H], Wi swizzled bf16, out fp32 [T*B,3H])
// One 16x16 tile per wave, K=1024 -> 32 WMMAs. 1024x192 tiles, 4 waves/block.
// ---------------------------------------------------------------------------
__global__ void gi_gemm_kernel(const bf16_t* __restrict__ x_bf,
                               const bf16_t* __restrict__ wi_sw,
                               const float* __restrict__ bi,
                               float* __restrict__ gi) {
  const int lane = threadIdx.x & 31;
  const int wave = threadIdx.x >> 5;
  const int tile = blockIdx.x * 4 + wave;          // < 1024*192
  const int rt = tile / 192;
  const int ct = tile % 192;
  const int rowA = rt * 16 + (lane & 15);
  const bf16_t* xr = x_bf + (size_t)rowA * Hn + ((lane >> 4) * 8);
  const bf16_t* wb = wi_sw + ((size_t)ct * 32 * 32 + lane) * 16;

  v8f acc = {};
#pragma unroll 4
  for (int kt = 0; kt < 32; ++kt) {
    v16bf a = make_a_frag_bf(xr + kt * 32);
    v16bf b = *(const v16bf*)(wb + (size_t)kt * (32 * 16));
    acc = wmma_bf16(a, b, acc);
  }
  const int n = ct * 16 + (lane & 15);
  const float biv = bi[n];
  const int mbase = rt * 16 + 8 * (lane >> 4);
#pragma unroll
  for (int i = 0; i < 8; ++i)
    gi[(size_t)(mbase + i) * H3 + n] = acc[i] + biv;
}

// ---------------------------------------------------------------------------
// Persistent GRU scan: 64 blocks x 128 threads (4 waves).
// Block owns h-columns [c0,c0+16); wave owns batch rows [16*wave,+16).
// fp32 carry (exact gates) + bf16 shadow (WMMA A operand), both double-buffered.
// Reset handled by pointer-select to a zero row (A side) and per-element
// select on the fp32 hold (z*h side).
// ---------------------------------------------------------------------------
__global__ void gru_scan_kernel(const float* __restrict__ gi,
                                const bf16_t* __restrict__ wh_sw,
                                const float* __restrict__ bn,
                                const int* __restrict__ resets,
                                float* __restrict__ hbuf0, float* __restrict__ hbuf1,
                                bf16_t* __restrict__ hb0,  bf16_t* __restrict__ hb1,
                                const bf16_t* __restrict__ zrow,
                                float* __restrict__ out_hfinal,
                                float* __restrict__ out_ys,
                                unsigned* __restrict__ sync) {
  const int lane = threadIdx.x & 31;
  const int wave = threadIdx.x >> 5;   // 0..3
  const int c0 = blockIdx.x * 16;      // h-column base (blockIdx.x < 64)
  const int b0 = wave * 16;            // batch-row base
  const int nIdx = lane & 15;
  const int base0 = (lane >> 4) * 8;

  // interleaved Wh fragments for this column block: 3KB per kt
  const bf16_t* wb = wh_sw + ((size_t)(blockIdx.x * 32 * 3) * 32 + lane) * 16;

  const float bnv = bn[c0 + nIdx];
  const int rowA = b0 + nIdx;          // A-fragment row for this lane

  float*  hc = hbuf0;  float*  hx = hbuf1;
  bf16_t* bc = hb0;    bf16_t* bx = hb1;

  for (int t = 0; t < Tn; ++t) {
    const int rstA = resets[t * Bn + rowA];
    const bf16_t* hA = rstA ? (zrow + base0)
                            : (bc + (size_t)rowA * Hn + base0);

    v8f ar = {}, az = {}, an = {};
#pragma unroll 4
    for (int kt = 0; kt < 32; ++kt) {
      v16bf a = make_a_frag_bf(hA + kt * 32);
      const bf16_t* wkt = wb + (size_t)kt * 1536;
      v16bf br = *(const v16bf*)(wkt);
      v16bf bz = *(const v16bf*)(wkt + 512);
      v16bf bg = *(const v16bf*)(wkt + 1024);
      ar = wmma_bf16(a, br, ar);
      az = wmma_bf16(a, bz, az);
      an = wmma_bf16(a, bg, an);
      __builtin_prefetch(wkt + 2 * 1536, 0, 0);   // global_prefetch on weight stream
    }

    // Elementwise gates. C/D layout: element i of lane l -> M = i + 8*(l>>4).
    const int mbase = b0 + base0;
#pragma unroll
    for (int i = 0; i < 8; ++i) {
      const int row = mbase + i;
      const size_t girow = ((size_t)t * Bn + row) * H3;
      const float ir  = gi[girow + 0 * Hn + c0 + nIdx];
      const float iz  = gi[girow + 1 * Hn + c0 + nIdx];
      const float inn = gi[girow + 2 * Hn + c0 + nIdx];
      float hold = hc[(size_t)row * Hn + c0 + nIdx];
      if (resets[t * Bn + row]) hold = 0.0f;
      const float r = sigf(ir + ar[i]);
      const float z = sigf(iz + az[i]);
      const float n = tanhfast(inn + r * (an[i] + bnv));
      const float hv = (1.0f - z) * n + z * hold;
      hx[(size_t)row * Hn + c0 + nIdx] = hv;
      bx[(size_t)row * Hn + c0 + nIdx] = (bf16_t)hv;
      out_ys[((size_t)t * Bn + row) * Hn + c0 + nIdx] = hv;
      if (t == Tn - 1) out_hfinal[(size_t)row * Hn + c0 + nIdx] = hv;
    }

    // ---- grid-wide barrier between timesteps ----
    __threadfence();
    __syncthreads();
    if (threadIdx.x == 0) {
      __hip_atomic_fetch_add(sync, 1u, __ATOMIC_ACQ_REL, __HIP_MEMORY_SCOPE_AGENT);
      const unsigned target = (unsigned)gridDim.x * (unsigned)(t + 1);
      while (__hip_atomic_load(sync, __ATOMIC_ACQUIRE, __HIP_MEMORY_SCOPE_AGENT) < target) {
        __builtin_amdgcn_s_sleep(1);
      }
    }
    __syncthreads();

    float*  tf = hc; hc = hx; hx = tf;
    bf16_t* tb = bc; bc = bx; bx = tb;
  }
}

// ---------------------------------------------------------------------------
// Host launch
// ---------------------------------------------------------------------------
extern "C" void kernel_launch(void* const* d_in, const int* in_sizes, int n_in,
                              void* d_out, int out_size, void* d_ws, size_t ws_size,
                              hipStream_t stream) {
  const float* x      = (const float*)d_in[0];   // [T,B,H]
  const int*   resets = (const int*)  d_in[1];   // [T,B]
  const float* h0     = (const float*)d_in[2];   // [B,H]
  const float* Wi     = (const float*)d_in[3];   // [H,3H]
  const float* bi     = (const float*)d_in[4];   // [3H]
  const float* Wh     = (const float*)d_in[5];   // [H,3H]
  const float* bn     = (const float*)d_in[6];   // [H]

  float* out_hfinal = (float*)d_out;                     // [B,H]
  float* out_ys     = (float*)d_out + (size_t)Bn * Hn;   // [T,B,H]

  // workspace layout
  const size_t GI_BYTES = (size_t)Tn * Bn * H3 * sizeof(float);     // 201.3 MB
  const size_t W_BYTES  = (size_t)Hn * H3 * sizeof(bf16_t);         // 6.29 MB each
  const size_t XB_BYTES = (size_t)Tn * Bn * Hn * sizeof(bf16_t);    // 33.6 MB
  const size_t H_BYTES  = (size_t)Bn * Hn * sizeof(float);          // 256 KB each
  const size_t HB_BYTES = (size_t)Bn * Hn * sizeof(bf16_t);         // 128 KB each
  char* wsp = (char*)d_ws;
  float*  gi_buf = (float*)wsp;    wsp += GI_BYTES;
  bf16_t* wi_sw  = (bf16_t*)wsp;   wsp += W_BYTES;
  bf16_t* wh_sw  = (bf16_t*)wsp;   wsp += W_BYTES;
  bf16_t* x_bf   = (bf16_t*)wsp;   wsp += XB_BYTES;
  float*  hbuf0  = (float*)wsp;    wsp += H_BYTES;
  float*  hbuf1  = (float*)wsp;    wsp += H_BYTES;
  bf16_t* hb0    = (bf16_t*)wsp;   wsp += HB_BYTES;
  bf16_t* hb1    = (bf16_t*)wsp;   wsp += HB_BYTES;
  bf16_t* zrow   = (bf16_t*)wsp;   wsp += (size_t)Hn * sizeof(bf16_t);
  unsigned* sync = (unsigned*)wsp; wsp += 128;
  if ((size_t)(wsp - (char*)d_ws) > ws_size) return;  // insufficient scratch

  // per-launch init (graph-replay safe)
  hipMemsetAsync(sync, 0, 128, stream);
  hipMemsetAsync(zrow, 0, (size_t)Hn * sizeof(bf16_t), stream);
  hipMemcpyAsync(hbuf0, h0, H_BYTES, hipMemcpyDeviceToDevice, stream);

  // 0) bf16 copies: x and h0 shadow
  {
    const int n8x = Tn * Bn * Hn / 8;
    cvt_bf16_kernel<<<(n8x + 255) / 256, 256, 0, stream>>>(x, x_bf, n8x);
    const int n8h = Bn * Hn / 8;
    cvt_bf16_kernel<<<(n8h + 255) / 256, 256, 0, stream>>>(h0, hb0, n8h);
  }

  // 1) swizzle weights into WMMA B-fragment order (bf16)
  {
    dim3 grid((192 * 32 * 32 + 255) / 256), block(256);
    swizzle_wi_kernel<<<grid, block, 0, stream>>>(Wi, wi_sw);
    swizzle_wh_kernel<<<grid, block, 0, stream>>>(Wh, wh_sw);
  }

  // 2) big parallel GEMM: Gi = X @ Wi + bi
  {
    const int tiles = (Tn * Bn / 16) * (H3 / 16);   // 196608
    gi_gemm_kernel<<<dim3(tiles / 4), dim3(128), 0, stream>>>(x_bf, wi_sw, bi, gi_buf);
  }

  // 3) persistent sequential scan with per-step grid barrier
  {
    gru_scan_kernel<<<dim3(64), dim3(128), 0, stream>>>(
        gi_buf, wh_sw, bn, resets, hbuf0, hbuf1, hb0, hb1, zrow,
        out_hfinal, out_ys, sync);
  }
}
